// QuantMoETorchFFN_85023172592039
// MI455X (gfx1250) — matrix operand order repaired and checked by reading, not compiled, not run on hardware
//
#include <hip/hip_runtime.h>
#include <hip/hip_bf16.h>

#define E_ 8
#define D_ 1024
#define H_ 4096
#define T_ 2048
#define TM 32      // tokens per block
#define HC 128     // H chunk columns
#define KSTEP 32   // WMMA K per step (f16)

typedef _Float16 v16h __attribute__((ext_vector_type(16)));
typedef _Float16 v8h  __attribute__((ext_vector_type(8)));
typedef float    v8f  __attribute__((ext_vector_type(8)));
typedef int      v4i  __attribute__((ext_vector_type(4)));
typedef int      v8i  __attribute__((ext_vector_type(8)));

__device__ __forceinline__ v8f v8f_zero() {
    v8f z;
#pragma unroll
    for (int i = 0; i < 8; ++i) z[i] = 0.0f;
    return z;
}

// A fragment (16x32 f16, M=tokens): lane L holds row M=L&15;
// lanes 0-15: K = k+0..15 (V0-7); lanes 16-31: K = k+16..31.
__device__ __forceinline__ v16h load_a_frag(const _Float16* row, int k, int lane) {
    int ko = k + ((lane >> 4) << 4);
    union { v16h v; v8h h[2]; } u;
    u.h[0] = *(const v8h*)(row + ko);        // ds_load_b128
    u.h[1] = *(const v8h*)(row + ko + 8);    // ds_load_b128
    return u.v;
}

// Raw B-fragment data: 16 consecutive int32 (q in 0..255) from one weight row.
__device__ __forceinline__ void load_b_raw(v4i r[4], const int* qptr) {
    const v4i* p = (const v4i*)qptr;
#pragma unroll
    for (int j = 0; j < 4; ++j) r[j] = p[j];  // 4x global_load_b128
}

// Bit-trick int->f16: for q in [0,255], bits (0x6400|q) == half(1024+q) exactly.
// Pack two ints per dword + OR the bias pattern: 2 VALU per dword, no cvt chain.
// The +1024 bias is folded into the zero-point correction in the epilogues.
__device__ __forceinline__ v16h cvt_b_frag(const v4i r[4]) {
    union { v8i i; v16h h; } u;
#pragma unroll
    for (int j = 0; j < 4; ++j) {
        u.i[2 * j]     = (r[j][0] | (r[j][1] << 16)) | 0x64006400;
        u.i[2 * j + 1] = (r[j][2] | (r[j][3] << 16)) | 0x64006400;
    }
    return u.h;
}

__device__ __forceinline__ v8f wmma_f16(v16h a, v16h b, v8f c) {
    return __builtin_amdgcn_wmma_f32_16x16x32_f16(false, a, false, b,
                                                  (short)0, c, false, false);
}

// ---------------------------------------------------------------------------
__global__ void moe_zero_kernel(float* __restrict__ out, int n,
                                int* __restrict__ counts) {
    int i = blockIdx.x * blockDim.x + threadIdx.x;
    if (i < n) out[i] = 0.0f;
    if (i < E_) counts[i] = 0;
}

// ---------------------------------------------------------------------------
// One wave per token: scores = x . wg^T, top-2, softmax, append to expert list
__global__ __launch_bounds__(256) void moe_gate_kernel(
    const float* __restrict__ xg, const float* __restrict__ wg,
    float* __restrict__ gates, int* __restrict__ counts, int* __restrict__ tidx) {
    int wave = threadIdx.x >> 5, lane = threadIdx.x & 31;
    int t = blockIdx.x * 8 + wave;
    if (t >= T_) return;

    float acc[E_];
#pragma unroll
    for (int e = 0; e < E_; ++e) acc[e] = 0.0f;
    for (int d = lane; d < D_; d += 32) {
        float xv = xg[(size_t)t * D_ + d];
#pragma unroll
        for (int e = 0; e < E_; ++e) acc[e] += xv * wg[e * D_ + d];
    }
#pragma unroll
    for (int e = 0; e < E_; ++e) {
#pragma unroll
        for (int o = 16; o > 0; o >>= 1) acc[e] += __shfl_xor(acc[e], o, 32);
    }
    if (lane == 0) {
        int e1 = 0;
#pragma unroll
        for (int e = 1; e < E_; ++e) if (acc[e] > acc[e1]) e1 = e;
        int e2 = (e1 == 0) ? 1 : 0;
#pragma unroll
        for (int e = 0; e < E_; ++e)
            if (e != e1 && acc[e] > acc[e2]) e2 = e;
        float p2 = __expf(acc[e2] - acc[e1]);
        float inv = 1.0f / (1.0f + p2);
        float g1 = inv, g2 = p2 * inv;
#pragma unroll
        for (int e = 0; e < E_; ++e)
            gates[(size_t)t * E_ + e] = (e == e1) ? g1 : ((e == e2) ? g2 : 0.0f);
        int p = atomicAdd(&counts[e1], 1); tidx[e1 * T_ + p] = t;
        int q = atomicAdd(&counts[e2], 1); tidx[e2 * T_ + q] = t;
    }
}

// ---------------------------------------------------------------------------
// grid = (T/TM, E). 8 waves. 32 gathered tokens x one expert per block.
__global__ __launch_bounds__(256) void moe_ffn_kernel(
    const float* __restrict__ xg,
    const int* __restrict__ q1, const float* __restrict__ s1, const float* __restrict__ z1,
    const int* __restrict__ q2, const float* __restrict__ s2, const float* __restrict__ z2,
    const int* __restrict__ q3, const float* __restrict__ s3, const float* __restrict__ z3,
    const float* __restrict__ gates, const int* __restrict__ counts,
    const int* __restrict__ tidx, float* __restrict__ out) {

    __shared__ _Float16 xs[TM][D_];   // 64 KB: gathered token rows (f16)
    __shared__ _Float16 act[TM][HC];  // 8 KB: SwiGLU chunk
    __shared__ int   tok[TM];
    __shared__ float gat[TM];
    __shared__ float xsum[TM];
    __shared__ float asum[TM];

    const int e = blockIdx.y;
    const int cnt = counts[e];
    const int base = blockIdx.x * TM;
    if (base >= cnt) return;

    const int tid = threadIdx.x;
    const int wave = tid >> 5;
    const int lane = tid & 31;

    if (tid < TM) {
        int i = base + tid;
        int t = (i < cnt) ? tidx[e * T_ + i] : tidx[e * T_ + cnt - 1];
        tok[tid] = t;
        gat[tid] = (i < cnt) ? gates[(size_t)t * E_ + e] : 0.0f;
        asum[tid] = 0.0f;
    }
    __syncthreads();

    // Gather x rows into LDS as f16 (vectorized b128 global loads)
    for (int v = tid; v < TM * (D_ / 4); v += 256) {
        int r = v / (D_ / 4);
        int c = (v % (D_ / 4)) * 4;
        float4 f = *(const float4*)(xg + (size_t)tok[r] * D_ + c);
        xs[r][c + 0] = (_Float16)f.x; xs[r][c + 1] = (_Float16)f.y;
        xs[r][c + 2] = (_Float16)f.z; xs[r][c + 3] = (_Float16)f.w;
    }
    __syncthreads();

    // Per-token row sums of f16 x (zero-point + bias correction of GEMM1/3)
    {
        int r0 = wave * (TM / 8);
#pragma unroll
        for (int rr = 0; rr < TM / 8; ++rr) {
            int r = r0 + rr;
            float s = 0.0f;
            for (int c = lane; c < D_; c += 32) s += (float)xs[r][c];
#pragma unroll
            for (int o = 16; o > 0; o >>= 1) s += __shfl_xor(s, o, 32);
            if (lane == 0) xsum[r] = s;
        }
    }
    __syncthreads();

    const size_t eHD = (size_t)e * H_ * D_;
    const size_t eDH = (size_t)e * D_ * H_;
    const int klaneoff = (lane >> 4) << 4;  // K half-group for B fragments
    const int mrow = lane & 15;             // A row, B/C column

    // Output accumulator: 32 tokens x 128 d-columns per wave (16 C tiles)
    v8f yacc[2][8];
#pragma unroll
    for (int mt = 0; mt < 2; ++mt)
#pragma unroll
        for (int nt = 0; nt < 8; ++nt) yacc[mt][nt] = v8f_zero();

    // Lane-local act row-sum partials: one per (mt, j) C-row; reduced once at end
    float lasum[16];
#pragma unroll
    for (int i = 0; i < 16; ++i) lasum[i] = 0.0f;

    // q2 row base pointers for this wave's 8 d-column tiles
    const int* q2base[8];
#pragma unroll
    for (int nt = 0; nt < 8; ++nt) {
        int dr = wave * 128 + nt * 16 + mrow;
        q2base[nt] = q2 + eDH + (size_t)dr * H_ + klaneoff;
    }

#pragma unroll 1
    for (int hc = 0; hc < H_; hc += HC) {
        // ---- GEMM1/GEMM3: this wave computes 16 h-columns of h1 and h3 ----
        int hr = hc + wave * 16 + mrow;     // weight row (h index) for this lane
        const int* q1row = q1 + eHD + (size_t)hr * D_ + klaneoff;
        const int* q3row = q3 + eHD + (size_t)hr * D_ + klaneoff;

        v8f acc1[2] = { v8f_zero(), v8f_zero() };
        v8f acc3[2] = { v8f_zero(), v8f_zero() };

        // Software-pipelined weight stream: load k+KSTEP raw while computing k
        v4i r1[4], r3[4];
        load_b_raw(r1, q1row);
        load_b_raw(r3, q3row);

#pragma unroll 2
        for (int k = 0; k < D_; k += KSTEP) {
            int kn = (k + KSTEP < D_) ? (k + KSTEP) : k;  // clamp (no OOB)
            v4i n1[4], n3[4];
            load_b_raw(n1, q1row + kn);
            load_b_raw(n3, q3row + kn);
            __builtin_prefetch(q1row + k + 8 * KSTEP, 0, 1);
            __builtin_prefetch(q3row + k + 8 * KSTEP, 0, 1);

            v16h b1 = cvt_b_frag(r1);
            v16h b3 = cvt_b_frag(r3);
#pragma unroll
            for (int mt = 0; mt < 2; ++mt) {
                v16h a = load_a_frag(&xs[mt * 16 + mrow][0], k, lane);
                acc1[mt] = wmma_f16(a, b1, acc1[mt]);
                acc3[mt] = wmma_f16(a, b3, acc3[mt]);
            }
#pragma unroll
            for (int j = 0; j < 4; ++j) { r1[j] = n1[j]; r3[j] = n3[j]; }
        }

        // Dequant epilogue + SwiGLU -> act chunk in LDS (bias 1024 folded in)
        float sc1 = s1[(size_t)e * H_ + hr], zo1 = z1[(size_t)e * H_ + hr] + 1024.0f;
        float sc3 = s3[(size_t)e * H_ + hr], zo3 = z3[(size_t)e * H_ + hr] + 1024.0f;
#pragma unroll
        for (int mt = 0; mt < 2; ++mt) {
#pragma unroll
            for (int j = 0; j < 8; ++j) {
                int m = mt * 16 + j + ((lane >> 4) << 3);  // token slot
                float h1v = sc1 * (acc1[mt][j] - zo1 * xsum[m]);
                float h3v = sc3 * (acc3[mt][j] - zo3 * xsum[m]);
                float av = (h1v / (1.0f + __expf(-h1v))) * h3v;
                act[m][wave * 16 + mrow] = (_Float16)av;
                lasum[mt * 8 + j] += av;                   // register, no LDS atomic
            }
        }
        __syncthreads();

        // ---- GEMM2: yacc += act_chunk @ q2_chunk^T ----
#pragma unroll
        for (int k = 0; k < HC; k += KSTEP) {
            v16h a0 = load_a_frag(&act[0 + mrow][0], k, lane);
            v16h a1 = load_a_frag(&act[16 + mrow][0], k, lane);
#pragma unroll
            for (int nt = 0; nt < 8; ++nt) {
                v4i rw[4];
                load_b_raw(rw, q2base[nt] + hc + k);
                v16h b = cvt_b_frag(rw);
                yacc[0][nt] = wmma_f16(a0, b, yacc[0][nt]);
                yacc[1][nt] = wmma_f16(a1, b, yacc[1][nt]);
            }
        }
        __syncthreads();  // act reused next chunk
    }

    // Reduce act row-sums: shuffle over the 16 column lanes, then one ds_add_f32
#pragma unroll
    for (int mt = 0; mt < 2; ++mt) {
#pragma unroll
        for (int j = 0; j < 8; ++j) {
            float v = lasum[mt * 8 + j];
#pragma unroll
            for (int o = 8; o > 0; o >>= 1) v += __shfl_xor(v, o, 32);
            if (mrow == 0)
                atomicAdd(&asum[mt * 16 + j + ((lane >> 4) << 3)], v);
        }
    }
    __syncthreads();

    // Final epilogue: y = gate * s2 * (acc - (z2+1024) * sum_h act); scatter-add
#pragma unroll
    for (int nt = 0; nt < 8; ++nt) {
        int d = wave * 128 + nt * 16 + mrow;
        float scl = s2[(size_t)e * D_ + d];
        float zo  = z2[(size_t)e * D_ + d] + 1024.0f;
#pragma unroll
        for (int mt = 0; mt < 2; ++mt) {
#pragma unroll
            for (int j = 0; j < 8; ++j) {
                int m = mt * 16 + j + ((lane >> 4) << 3);
                float y = gat[m] * scl * (yacc[mt][nt][j] - zo * asum[m]);
                atomicAdd(out + (size_t)tok[m] * D_ + d, y);
            }
        }
    }
}

// ---------------------------------------------------------------------------
extern "C" void kernel_launch(void* const* d_in, const int* in_sizes, int n_in,
                              void* d_out, int out_size, void* d_ws, size_t ws_size,
                              hipStream_t stream) {
    const float* x  = (const float*)d_in[0];
    const float* wg = (const float*)d_in[1];
    const int*   q1 = (const int*)d_in[2];
    const float* s1 = (const float*)d_in[3];
    const float* z1 = (const float*)d_in[4];
    const int*   q2 = (const int*)d_in[5];
    const float* s2 = (const float*)d_in[6];
    const float* z2 = (const float*)d_in[7];
    const int*   q3 = (const int*)d_in[8];
    const float* s3 = (const float*)d_in[9];
    const float* z3 = (const float*)d_in[10];
    float* out = (float*)d_out;

    char* ws = (char*)d_ws;
    float* gates = (float*)ws;                       // T*E f32 = 64 KB
    int* counts  = (int*)(ws + 65536);               // E ints (padded 256B)
    int* tidx    = (int*)(ws + 65536 + 256);         // E*T ints = 64 KB

    int n = T_ * D_;
    moe_zero_kernel<<<(n + 255) / 256, 256, 0, stream>>>(out, n, counts);
    moe_gate_kernel<<<T_ / 8, 256, 0, stream>>>(x, wg, gates, counts, tidx);
    dim3 grid(T_ / TM, E_);
    moe_ffn_kernel<<<grid, 256, 0, stream>>>(x, q1, s1, z1, q2, s2, z2,
                                             q3, s3, z3, gates, counts, tidx, out);
}
